// DenseFAGCNConv_9457517986401
// MI455X (gfx1250) — compile-verified, matrix operand order checked by reading, not run on hardware
//
#include <hip/hip_runtime.h>
#include <hip/hip_bf16.h>
#include <math.h>

// ---------------------------------------------------------------------------
// DenseFAGCNConv on MI455X (gfx1250, wave32, WMMA, async-to-LDS).
//   B=8, N=2048, Cin=Cout=128.
//   Memory floor: adj = 134 MB -> ~5.8 us @ 23.3 TB/s.
//   Compute: out = (mask(adj) * tanh(ar al^T)) @ h = 8.6 GFLOP -> bf16 WMMA.
//   alpha is never materialized in HBM; h staged to LDS via async DMA.
// ---------------------------------------------------------------------------

typedef __attribute__((ext_vector_type(16))) __bf16 v16bf;
typedef __attribute__((ext_vector_type(8)))  float  v8f;
typedef int v4i_vs __attribute__((__vector_size__(16)));   // matches builtin param

#define BDIM 8
#define NDIM 2048
#define CDIM 128
#define EPSV 0.1f

#define KB      64   // K elems staged per pipeline step
#define HSTRIDE 72   // hT row stride (bf16): 64 data + 8 pad = 144B (16B-mult, conflict-free)
#define ASTRIDE 68   // aT row stride (bf16): 136B = 34 dwords (conflict-free)
#define LPAD    34   // k_linear tile stride

// ---- CDNA5 feature detection (device pass only; host pass takes fallbacks) --
#if defined(__has_builtin)
#  if __has_builtin(__builtin_amdgcn_global_load_async_to_lds_b128) && \
      __has_builtin(__builtin_amdgcn_s_wait_asynccnt)
#    define HAVE_ASYNC 1
#  endif
#  if __has_builtin(__builtin_amdgcn_tanhf)
#    define HAVE_TANH 1
#  endif
#endif
#ifndef HAVE_ASYNC
#define HAVE_ASYNC 0
#endif
#ifndef HAVE_TANH
#define HAVE_TANH 0
#endif

__device__ __forceinline__ float fast_tanh(float v) {
#if HAVE_TANH
    return __builtin_amdgcn_tanhf(v);     // v_tanh_f32
#else
    return tanhf(v);
#endif
}

__device__ __forceinline__ void async_wait_all() {
#if HAVE_ASYNC
    __builtin_amdgcn_s_wait_asynccnt(0);  // s_wait_asynccnt 0
#endif
}

// 16-byte chunk: global -> LDS (async DMA if available)
__device__ __forceinline__ void copy16_g2l(const void* gsrc, void* ldst) {
#if HAVE_ASYNC
    __builtin_amdgcn_global_load_async_to_lds_b128(
        (__attribute__((address_space(1))) v4i_vs*)(gsrc),
        (__attribute__((address_space(3))) v4i_vs*)(ldst), 0, 0);
#else
    *(uint4*)ldst = *(const uint4*)gsrc;
#endif
}

// ---------------------------------------------------------------------------
// Kernel 1: u_l = W @ w_att_l, u_r = W @ w_att_r (exact f32) so that
// alpha_l = x @ u_l matches the reference's f32 h @ w_att path.
// ---------------------------------------------------------------------------
__global__ void k_u(const float* __restrict__ W, const float* __restrict__ wl,
                    const float* __restrict__ wr, float* __restrict__ ul,
                    float* __restrict__ ur) {
    int i = threadIdx.x;  // 128 threads
    float sl = 0.f, sr = 0.f;
    for (int o = 0; o < CDIM; ++o) {
        float w = W[i * CDIM + o];
        sl += w * wl[o];
        sr += w * wr[o];
    }
    ul[i] = sl;
    ur[i] = sr;
}

// ---------------------------------------------------------------------------
// Kernel 2: h = x @ W -> bf16, stored TRANSPOSED as hbt[b][c][n] so the
// aggregation kernel can async-DMA contiguous 128B rows straight into LDS.
// ---------------------------------------------------------------------------
__global__ __launch_bounds__(256) void k_linear(const float* __restrict__ x,
                                                const float* __restrict__ W,
                                                __bf16* __restrict__ hbt) {
    __shared__ __bf16 xT[16 * LPAD];    // A tile [m][k]
    __shared__ __bf16 WT[128 * LPAD];   // B tile transposed [c][k]
    const int b = blockIdx.y;
    const int i0 = blockIdx.x * 16;
    const int tid = threadIdx.x;
    const int wave = tid >> 5, lane = tid & 31;
    const int c0 = wave * 16;
    const long xBase = (long)b * NDIM * CDIM;

    v8f acc = {};
    for (int k0 = 0; k0 < CDIM; k0 += 32) {
        __syncthreads();
#pragma unroll
        for (int q = 0; q < 2; ++q) {
            int e = tid + q * 256;
            int m = e >> 5, k = e & 31;
            xT[m * LPAD + k] = (__bf16)x[xBase + (long)(i0 + m) * CDIM + k0 + k];
        }
#pragma unroll
        for (int q = 0; q < 16; ++q) {
            int e = tid + q * 256;
            int kk = e >> 7, c = e & 127;  // coalesced over c
            WT[c * LPAD + kk] = (__bf16)W[(k0 + kk) * CDIM + c];
        }
        __syncthreads();

        const int n = lane & 15, g = lane >> 4;
        v16bf af, bfb;
#pragma unroll
        for (int v = 0; v < 8; ++v)
#pragma unroll
            for (int t = 0; t < 2; ++t) {
                int K = 16 * (v >> 2) + 2 * (v & 3) + t + 8 * g;
                af[2 * v + t]  = xT[(lane & 15) * LPAD + K];
                bfb[2 * v + t] = WT[(c0 + n) * LPAD + K];
            }
        acc = __builtin_amdgcn_wmma_f32_16x16x32_bf16(false, af, false, bfb,
                                                      (short)0, acc, false, false);
    }

    const int n = lane & 15, g = lane >> 4;
#pragma unroll
    for (int v = 0; v < 8; ++v) {        // transposed store: hbt[b][c0+n][row]
        int row = i0 + v + 8 * g;
        hbt[(long)b * CDIM * NDIM + (long)(c0 + n) * NDIM + row] = (__bf16)acc[v];
    }
}

// ---------------------------------------------------------------------------
// Kernel 3: alpha_l[b,i] = x[b,i,:] . u_l ; alpha_r likewise. One wave/row.
// ---------------------------------------------------------------------------
__global__ __launch_bounds__(256) void k_alpha(const float* __restrict__ x,
                                               const float* __restrict__ ul,
                                               const float* __restrict__ ur,
                                               float* __restrict__ al,
                                               float* __restrict__ ar) {
    __shared__ float uls[CDIM], urs[CDIM];
    const int tid = threadIdx.x;
    if (tid < 128) uls[tid] = ul[tid];
    else           urs[tid - 128] = ur[tid - 128];
    __syncthreads();

    const int row  = blockIdx.x * 8 + (tid >> 5);  // flat b*N+i
    const int lane = tid & 31;
    const float* xr = x + (long)row * CDIM;
    float sl = 0.f, sr = 0.f;
#pragma unroll
    for (int q = 0; q < 4; ++q) {
        int c = lane + q * 32;
        float v = xr[c];
        sl += v * uls[c];
        sr += v * urs[c];
    }
#pragma unroll
    for (int m = 16; m >= 1; m >>= 1) {
        sl += __shfl_xor(sl, m, 32);
        sr += __shfl_xor(sr, m, 32);
    }
    if (lane == 0) { al[row] = sl; ar[row] = sr; }
}

// ---------------------------------------------------------------------------
// Kernel 4: out = (mask(adj)*tanh(ar al^T)) @ h + EPS*x_0.
// WG = 16 rows x 128 cols, 8 waves. Double-buffered pipeline:
//   issue async h-stage(next) + build masked-tanh A(next)  ||  WMMA(cur)
// 2 x v_wmma_f32_16x16x32_bf16 per wave per K=64 step, 32 steps.
// ---------------------------------------------------------------------------
__global__ __launch_bounds__(256) void k_aggregate(const int* __restrict__ adj,
                                                   const float* __restrict__ x0,
                                                   const float* __restrict__ al,
                                                   const float* __restrict__ ar,
                                                   const __bf16* __restrict__ hbt,
                                                   float* __restrict__ out) {
    __shared__ __attribute__((aligned(16))) __bf16 hT[2][128 * HSTRIDE];
    __shared__ __attribute__((aligned(16))) __bf16 aT[2][16 * ASTRIDE];

    const int b = blockIdx.y;
    const int i0 = blockIdx.x * 16;
    const int tid = threadIdx.x;
    const int wave = tid >> 5, lane = tid & 31;
    const int c0 = wave * 16;
    const long adjBase = (long)b * NDIM * NDIM;
    const __bf16* hbase = hbt + (long)b * CDIM * NDIM;  // row c at + c*N

    // This thread builds A rows {mrow, mrow+4, mrow+8, mrow+12}, column kk.
    const int mrow = tid >> 6;       // 0..3
    const int kk   = tid & 63;       // 0..63
    float arv[4];
#pragma unroll
    for (int q = 0; q < 4; ++q) arv[q] = ar[b * NDIM + i0 + mrow + q * 4];

    // ---- pipeline stage helpers -------------------------------------------
    auto stage_h = [&](int j0, int buf) {       // 128 rows x 128B -> LDS (async)
#pragma unroll
        for (int q = 0; q < 4; ++q) {
            int p = tid + q * 256;              // 0..1023 16B-chunks
            int c = p >> 3, chunk = p & 7;
            copy16_g2l((const char*)(hbase + (long)c * NDIM + j0) + chunk * 16,
                       (char*)&hT[buf][c * HSTRIDE] + chunk * 16);
        }
    };
    auto build_a = [&](int j0, int buf) {       // masked tanh(ar*al) -> bf16 LDS
        float alv = al[b * NDIM + j0 + kk];
#pragma unroll
        for (int q = 0; q < 4; ++q) {
            int m = mrow + q * 4;
            int av = adj[adjBase + (long)(i0 + m) * NDIM + j0 + kk];
            float a = (av != 0) ? fast_tanh(arv[q] * alv) : 0.f;
            aT[buf][m * ASTRIDE + kk] = (__bf16)a;
        }
        if (j0 + KB < NDIM)                     // stream next adj chunk
            __builtin_prefetch(&adj[adjBase + (long)(i0 + mrow) * NDIM + j0 + KB + kk], 0, 1);
    };

    // ---- prologue ----------------------------------------------------------
    stage_h(0, 0);
    build_a(0, 0);
    async_wait_all();
    __syncthreads();

    v8f acc = {};
    const int n = lane & 15, g = lane >> 4;
    for (int it = 0; it < NDIM / KB; ++it) {
        const int cur = it & 1, nxt = cur ^ 1;
        const int j0n = (it + 1) * KB;
        if (j0n < NDIM) {                       // overlap: DMA + A-build vs WMMA
            stage_h(j0n, nxt);
            build_a(j0n, nxt);
        }
#pragma unroll
        for (int kc = 0; kc < 2; ++kc) {        // consume cur: K = 64
            v16bf af, bfb;
#pragma unroll
            for (int v = 0; v < 8; ++v)
#pragma unroll
                for (int t = 0; t < 2; ++t) {
                    int K = kc * 32 + 16 * (v >> 2) + 2 * (v & 3) + t + 8 * g;
                    af[2 * v + t]  = aT[cur][(lane & 15) * ASTRIDE + K];
                    bfb[2 * v + t] = hT[cur][(c0 + n) * HSTRIDE + K];
                }
            acc = __builtin_amdgcn_wmma_f32_16x16x32_bf16(false, af, false, bfb,
                                                          (short)0, acc, false, false);
        }
        async_wait_all();                       // next buffer fully landed
        __syncthreads();                        // visible to all waves; cur reusable
    }

    // ---- epilogue: out is row-major [B][N][C] ------------------------------
#pragma unroll
    for (int v = 0; v < 8; ++v) {
        int row = i0 + v + 8 * g;
        long idx = (long)b * NDIM * CDIM + (long)row * CDIM + c0 + n;
        out[idx] = acc[v] + EPSV * x0[idx];
    }
}

// ---------------------------------------------------------------------------
extern "C" void kernel_launch(void* const* d_in, const int* in_sizes, int n_in,
                              void* d_out, int out_size, void* d_ws, size_t ws_size,
                              hipStream_t stream) {
    const float* x   = (const float*)d_in[0];
    const float* x0  = (const float*)d_in[1];
    const int*   adj = (const int*)d_in[2];
    const float* W   = (const float*)d_in[3];
    const float* wl  = (const float*)d_in[4];
    const float* wr  = (const float*)d_in[5];
    float* out = (float*)d_out;

    char* ws = (char*)d_ws;
    __bf16* hbt = (__bf16*)ws;                                  // 4 MB, [B][C][N]
    float*  al  = (float*)(ws + 4u * 1024 * 1024);              // 64 KB
    float*  ar  = (float*)(ws + 4u * 1024 * 1024 + 64 * 1024);  // 64 KB
    float*  ul  = (float*)(ws + 4u * 1024 * 1024 + 128 * 1024); // 512 B
    float*  ur  = ul + CDIM;

    k_u<<<1, 128, 0, stream>>>(W, wl, wr, ul, ur);
    k_linear<<<dim3(NDIM / 16, BDIM), 256, 0, stream>>>(x, W, hbt);
    k_alpha<<<(BDIM * NDIM) / 8, 256, 0, stream>>>(x, ul, ur, al, ar);
    k_aggregate<<<dim3(NDIM / 16, BDIM), 256, 0, stream>>>(adj, x0, al, ar, hbt, out);
}
